// DeepSeekMoE_36258113913354
// MI455X (gfx1250) — compile-verified, matrix operand order checked by reading, not run on hardware
//
#include <hip/hip_runtime.h>
#include <hip/hip_bf16.h>
#include <math.h>

// ---------------------------------------------------------------------------
// DeepSeek-style MoE (top-2 of 8 experts + shared expert) for gfx1250.
// f16 WMMA (v_wmma_f32_16x16x32_f16), f32 accumulate. Compute-bound problem
// (~600 FLOP/byte >> ~107 FLOP/byte machine balance). Only top-2 experts are
// computed (grouped GEMM over per-expert gathered token lists). Tile staging
// uses CDNA5 async global->LDS copies (GLOBAL_LOAD_ASYNC_TO_LDS_B128,
// ASYNCcnt-tracked) with double-buffered LDS and one barrier per K-step.
// ---------------------------------------------------------------------------

typedef __attribute__((ext_vector_type(16))) _Float16 v16h;
typedef __attribute__((ext_vector_type(8)))  _Float16 v8h;
typedef __attribute__((ext_vector_type(8)))  float    v8f;

#define T_TOK 4096
#define D_DIM 1024
#define H_HID 2048
#define N_EXP 8

#define BM 128
#define BN 128
#define KC 32
#define KPAD 40  // KC + 8 halfs of padding (80B row stride, 16B aligned)

union H16frag { v16h v; v8h h[2]; };

__device__ __forceinline__ float gelu_exact(float x) {
    // exact (erf) GELU, matches jax.nn.gelu(approximate=False)
    return 0.5f * x * (1.0f + erff(x * 0.70710678118654752f));
}

// CDNA5 async copy: 16B per lane, global -> LDS, tracked by ASYNCcnt.
__device__ __forceinline__ void async_copy_b128(unsigned lds_byte_addr, const void* gaddr) {
    asm volatile("global_load_async_to_lds_b128 %0, %1, off"
                 :: "v"(lds_byte_addr), "v"(gaddr) : "memory");
}
__device__ __forceinline__ void wait_asynccnt0() {
    asm volatile("s_wait_asynccnt 0x0" ::: "memory");
}
// Generic->LDS byte offset: low 32 bits of the generic pointer are the LDS
// offset; hardware adds the wave's LDS_BASE (ISA 10.2 LDS addressing).
__device__ __forceinline__ unsigned lds_addr_of(const void* p) {
    return (unsigned)(unsigned long long)p;
}

// ---------------------------------------------------------------------------
// Elementwise f32 -> f16 convert (activations)
// ---------------------------------------------------------------------------
__global__ __launch_bounds__(256) void conv_f16_kernel(const float* __restrict__ in,
                                                       _Float16* __restrict__ out, int n) {
    int i = blockIdx.x * 256 + threadIdx.x;
    if (i < n) out[i] = (_Float16)in[i];
}

// ---------------------------------------------------------------------------
// Tiled transpose + convert: in [z][R][C] f32 -> out [z][C][R] f16
// (weights become [N][K] row-major so B-fragments load like A-fragments)
// ---------------------------------------------------------------------------
__global__ __launch_bounds__(256) void transpose_f16_kernel(const float* __restrict__ in,
                                                            _Float16* __restrict__ out,
                                                            int R, int C) {
    __shared__ float tile[32][33];
    size_t zo = (size_t)blockIdx.z * R * C;
    int r0 = blockIdx.y * 32, c0 = blockIdx.x * 32;
    int tx = threadIdx.x, ty = threadIdx.y;  // blockDim = (32, 8)
#pragma unroll
    for (int i = 0; i < 32; i += 8) {
        int r = r0 + ty + i, c = c0 + tx;
        if (r < R && c < C) tile[ty + i][tx] = in[zo + (size_t)r * C + c];
    }
    __syncthreads();
#pragma unroll
    for (int i = 0; i < 32; i += 8) {
        int c = c0 + ty + i, r = r0 + tx;
        if (r < R && c < C) out[zo + (size_t)c * R + r] = (_Float16)tile[tx][ty + i];
    }
}

// ---------------------------------------------------------------------------
// Router: one wave (32 lanes) per token. fp32 logits -> softmax -> top2.
// ---------------------------------------------------------------------------
__global__ __launch_bounds__(256) void router_kernel(const float* __restrict__ x,
                                                     const float* __restrict__ rw,
                                                     const float* __restrict__ rb,
                                                     int* __restrict__ top_e,
                                                     float* __restrict__ top_w) {
    int lane = threadIdx.x & 31;
    int t = blockIdx.x * 8 + (threadIdx.x >> 5);
    if (t >= T_TOK) return;
    const float* xr = x + (size_t)t * D_DIM;
    float a[N_EXP];
#pragma unroll
    for (int e = 0; e < N_EXP; ++e) a[e] = 0.0f;
    for (int i = lane; i < D_DIM; i += 32) {
        float xv = xr[i];
        const float* wr = rw + (size_t)i * N_EXP;
#pragma unroll
        for (int e = 0; e < N_EXP; ++e) a[e] += xv * wr[e];
    }
#pragma unroll
    for (int off = 16; off > 0; off >>= 1)
#pragma unroll
        for (int e = 0; e < N_EXP; ++e) a[e] += __shfl_down(a[e], off, 32);
    if (lane == 0) {
        float p[N_EXP];
        float mx = -1e30f;
#pragma unroll
        for (int e = 0; e < N_EXP; ++e) { p[e] = a[e] + rb[e]; mx = fmaxf(mx, p[e]); }
        float s = 0.0f;
#pragma unroll
        for (int e = 0; e < N_EXP; ++e) { p[e] = __expf(p[e] - mx); s += p[e]; }
        float inv = 1.0f / s;
#pragma unroll
        for (int e = 0; e < N_EXP; ++e) p[e] *= inv;
        int i0 = 0;
#pragma unroll
        for (int e = 1; e < N_EXP; ++e) if (p[e] > p[i0]) i0 = e;
        int i1 = (i0 == 0) ? 1 : 0;
#pragma unroll
        for (int e = 0; e < N_EXP; ++e) if (e != i0 && p[e] > p[i1]) i1 = e;
        float z = p[i0] + p[i1];
        top_e[t * 2 + 0] = i0; top_e[t * 2 + 1] = i1;
        top_w[t * 2 + 0] = p[i0] / z; top_w[t * 2 + 1] = p[i1] / z;
    }
}

// ---------------------------------------------------------------------------
// Per-expert counts + exclusive prefix (single block; E = 8, 2T = 8192 slots)
// ---------------------------------------------------------------------------
__global__ __launch_bounds__(256) void count_prefix_kernel(const int* __restrict__ top_e,
                                                           int* __restrict__ ex_base,
                                                           int* __restrict__ ex_cnt,
                                                           int* __restrict__ cursor) {
    __shared__ int c[N_EXP];
    int tid = threadIdx.x;
    if (tid < N_EXP) c[tid] = 0;
    __syncthreads();
    for (int s = tid; s < 2 * T_TOK; s += 256) atomicAdd(&c[top_e[s]], 1);
    __syncthreads();
    if (tid == 0) {
        int b = 0;
        for (int e = 0; e < N_EXP; ++e) { ex_base[e] = b; ex_cnt[e] = c[e]; cursor[e] = 0; b += c[e]; }
    }
}

// Assign each (token, k) pair a compact slot within its expert's range.
// Slot order is non-deterministic but every slot's VALUE is a pure function of
// its token, so final output values are deterministic.
__global__ __launch_bounds__(256) void scatter_kernel(const int* __restrict__ top_e,
                                                      const int* __restrict__ ex_base,
                                                      int* __restrict__ cursor,
                                                      int* __restrict__ slot_tok,
                                                      int* __restrict__ tok_slot) {
    int s = blockIdx.x * 256 + threadIdx.x;
    if (s >= 2 * T_TOK) return;
    int e = top_e[s];
    int local = atomicAdd(&cursor[e], 1);
    int slot = ex_base[e] + local;
    slot_tok[slot] = s >> 1;
    tok_slot[s] = slot;
}

// ---------------------------------------------------------------------------
// WMMA GEMM: C = act(A[M,K] @ Bt[N,K]^T + bias).
// 256 threads = 8 waves, block tile 128x128, K step 32, double-buffered LDS.
// Tiles staged with GLOBAL_LOAD_ASYNC_TO_LDS_B128 (ASYNCcnt); the next tile's
// loads are in flight while the current tile feeds the WMMA pipe; one
// s_wait_asynccnt 0 + workgroup barrier per K-step.
// Wave (wm in 0..1, wn in 0..3) owns a 64x32 patch = 4x2 frags of 16x16.
//   ACT_GELU: apply exact GELU at store
//   OUT_H   : store f16 (hidden activations) vs f32
//   EXPERT_M: blockIdx.z = expert; row range = [ex_base[e], +ex_cnt[e])
//   GATHER_A: A rows indirected through slot_tok (expert GEMM1 gathers tokens)
// ---------------------------------------------------------------------------
template <bool ACT_GELU, bool OUT_H, bool EXPERT_M, bool GATHER_A>
__global__ __launch_bounds__(256) void gemm_f16_kernel(
    const _Float16* __restrict__ A, const _Float16* __restrict__ Bt,
    const float* __restrict__ bias, float* __restrict__ Cf, _Float16* __restrict__ Ch,
    int M, int N, int K,
    const int* __restrict__ ex_base, const int* __restrict__ ex_cnt,
    const int* __restrict__ slot_tok) {

    __shared__ __align__(16) _Float16 As[2][BM][KPAD];
    __shared__ __align__(16) _Float16 Bs[2][BN][KPAD];

    const int e = EXPERT_M ? blockIdx.z : 0;
    const int base = EXPERT_M ? ex_base[e] : 0;
    const int Mrows = EXPERT_M ? ex_cnt[e] : M;
    const int m0 = blockIdx.y * BM;
    if (m0 >= Mrows) return;                     // block-uniform: safe pre-barrier exit
    const int n0 = blockIdx.x * BN;

    const _Float16* Bp = Bt + (EXPERT_M ? (size_t)e * N * K : 0);
    const float* biasp = bias + (EXPERT_M ? (size_t)e * N : 0);

    const int tid   = threadIdx.x;
    const int lane  = tid & 31;
    const int wid   = tid >> 5;
    const int wm    = wid >> 2;      // 0..1  -> 64-row half
    const int wn    = wid & 3;       // 0..3  -> 32-col quarter
    const int frow  = lane & 15;     // row (A) / col (B) within frag
    const int khalf = lane >> 4;     // K-half select for 16-bit frags

    // Stage one 128x32 K-slice of A and B into LDS buffer `buf` via async
    // copies (16B per lane per op). Rows past Mrows are zero-filled with a
    // plain ds_store; both paths complete before the end-of-step barrier.
    auto stage = [&](int buf, int kt) {
#pragma unroll
        for (int c = tid; c < (BM * KC) / 8; c += 256) {   // 512 chunks, 2 iters
            int row = c >> 2, col = (c & 3) << 3;
            int m = m0 + row;
            if (m < Mrows) {
                size_t arow = GATHER_A ? (size_t)slot_tok[base + m]
                            : (EXPERT_M ? (size_t)(base + m) : (size_t)m);
                async_copy_b128(lds_addr_of(&As[buf][row][col]), A + arow * K + kt + col);
            } else {
                v8h z = {};
                *(v8h*)&As[buf][row][col] = z;
            }
        }
#pragma unroll
        for (int c = tid; c < (BN * KC) / 8; c += 256) {   // N always multiple of 128
            int row = c >> 2, col = (c & 3) << 3;
            async_copy_b128(lds_addr_of(&Bs[buf][row][col]),
                            Bp + (size_t)(n0 + row) * K + kt + col);
        }
    };

    v8f cf[4][2];
#pragma unroll
    for (int i = 0; i < 4; ++i)
#pragma unroll
        for (int j = 0; j < 2; ++j) cf[i][j] = {};

    stage(0, 0);
    wait_asynccnt0();
    __syncthreads();

    int cur = 0;
    for (int kt = 0; kt < K; kt += KC) {
        if (kt + KC < K) stage(cur ^ 1, kt + KC);  // overlap next tile with compute

        // ---- fragments per ISA 16-bit layout: two 16B chunks per lane ----
        H16frag bfr[2];
#pragma unroll
        for (int fn = 0; fn < 2; ++fn) {
            const _Float16* p = &Bs[cur][wn * 32 + fn * 16 + frow][0];
            bfr[fn].h[0] = *(const v8h*)(p + khalf * 8);
            bfr[fn].h[1] = *(const v8h*)(p + 16 + khalf * 8);
        }
#pragma unroll
        for (int fm = 0; fm < 4; ++fm) {
            H16frag afr;
            const _Float16* p = &As[cur][wm * 64 + fm * 16 + frow][0];
            afr.h[0] = *(const v8h*)(p + khalf * 8);
            afr.h[1] = *(const v8h*)(p + 16 + khalf * 8);
#pragma unroll
            for (int fn = 0; fn < 2; ++fn)
                cf[fm][fn] = __builtin_amdgcn_wmma_f32_16x16x32_f16(
                    false, afr.v, false, bfr[fn].v, (short)0, cf[fm][fn], false, false);
        }

        wait_asynccnt0();   // my staged (next-tile) async copies are in LDS
        __syncthreads();    // everyone done computing `cur` and staging `cur^1`
        cur ^= 1;
    }

    // ---- epilogue: bias + optional GELU; C/D frag: elem r -> M = r + 8*(lane>=16)
    const int ncol = lane & 15;
    const int moff = (lane >> 4) * 8;
#pragma unroll
    for (int fm = 0; fm < 4; ++fm)
#pragma unroll
        for (int fn = 0; fn < 2; ++fn) {
            int nglob = n0 + wn * 32 + fn * 16 + ncol;
            float bv = biasp[nglob];
#pragma unroll
            for (int r = 0; r < 8; ++r) {
                int m = m0 + wm * 64 + fm * 16 + moff + r;
                if (m < Mrows) {
                    float v = cf[fm][fn][r] + bv;
                    if (ACT_GELU) v = gelu_exact(v);
                    size_t crow = EXPERT_M ? (size_t)(base + m) : (size_t)m;
                    if (OUT_H) Ch[crow * N + nglob] = (_Float16)v;
                    else       Cf[crow * N + nglob] = v;
                }
            }
        }
}

// ---------------------------------------------------------------------------
// Final combine: out = shared + w0*eout[slot0] + w1*eout[slot1] (deterministic)
// ---------------------------------------------------------------------------
__global__ __launch_bounds__(256) void combine_kernel(float* __restrict__ out,
                                                      const float* __restrict__ eout,
                                                      const float* __restrict__ top_w,
                                                      const int* __restrict__ tok_slot) {
    int idx = blockIdx.x * 256 + threadIdx.x;
    if (idx >= T_TOK * D_DIM) return;
    int t = idx >> 10, d = idx & (D_DIM - 1);
    int s0 = tok_slot[t * 2 + 0], s1 = tok_slot[t * 2 + 1];
    out[idx] += top_w[t * 2 + 0] * eout[(size_t)s0 * D_DIM + d]
              + top_w[t * 2 + 1] * eout[(size_t)s1 * D_DIM + d];
}

// ---------------------------------------------------------------------------
// Host launcher
// ---------------------------------------------------------------------------
extern "C" void kernel_launch(void* const* d_in, const int* in_sizes, int n_in,
                              void* d_out, int out_size, void* d_ws, size_t ws_size,
                              hipStream_t stream) {
    const float* x   = (const float*)d_in[0];
    const float* rw  = (const float*)d_in[1];
    const float* rb  = (const float*)d_in[2];
    const float* sw1 = (const float*)d_in[3];
    const float* sb1 = (const float*)d_in[4];
    const float* sw2 = (const float*)d_in[5];
    const float* sb2 = (const float*)d_in[6];
    const float* ew1 = (const float*)d_in[7];
    const float* eb1 = (const float*)d_in[8];
    const float* ew2 = (const float*)d_in[9];
    const float* eb2 = (const float*)d_in[10];
    float* out = (float*)d_out;

    // workspace carve-out (256B aligned); total ~160 MB
    char* w = (char*)d_ws;
    auto carve = [&](size_t bytes) -> char* {
        char* p = w; w += (bytes + 255) & ~(size_t)255; return p;
    };
    _Float16* xb    = (_Float16*)carve((size_t)T_TOK * D_DIM * 2);
    _Float16* sw1t  = (_Float16*)carve((size_t)H_HID * D_DIM * 2);
    _Float16* sw2t  = (_Float16*)carve((size_t)D_DIM * H_HID * 2);
    _Float16* ew1t  = (_Float16*)carve((size_t)N_EXP * H_HID * D_DIM * 2);
    _Float16* ew2t  = (_Float16*)carve((size_t)N_EXP * D_DIM * H_HID * 2);
    _Float16* hsh   = (_Float16*)carve((size_t)T_TOK * H_HID * 2);
    _Float16* hex   = (_Float16*)carve((size_t)2 * T_TOK * H_HID * 2);
    float*    eout  = (float*)   carve((size_t)2 * T_TOK * D_DIM * 4);
    int*      top_e = (int*)     carve((size_t)T_TOK * 2 * 4);
    float*    top_w = (float*)   carve((size_t)T_TOK * 2 * 4);
    int*      ex_base  = (int*)  carve(N_EXP * 4);
    int*      ex_cnt   = (int*)  carve(N_EXP * 4);
    int*      cursor   = (int*)  carve(N_EXP * 4);
    int*      slot_tok = (int*)  carve((size_t)2 * T_TOK * 4);
    int*      tok_slot = (int*)  carve((size_t)2 * T_TOK * 4);

    dim3 blk256(256);

    // 1) precision staging
    conv_f16_kernel<<<(T_TOK * D_DIM + 255) / 256, blk256, 0, stream>>>(x, xb, T_TOK * D_DIM);
    dim3 tb(32, 8);
    transpose_f16_kernel<<<dim3(H_HID / 32, D_DIM / 32, 1),     tb, 0, stream>>>(sw1, sw1t, D_DIM, H_HID);
    transpose_f16_kernel<<<dim3(D_DIM / 32, H_HID / 32, 1),     tb, 0, stream>>>(sw2, sw2t, H_HID, D_DIM);
    transpose_f16_kernel<<<dim3(H_HID / 32, D_DIM / 32, N_EXP), tb, 0, stream>>>(ew1, ew1t, D_DIM, H_HID);
    transpose_f16_kernel<<<dim3(D_DIM / 32, H_HID / 32, N_EXP), tb, 0, stream>>>(ew2, ew2t, H_HID, D_DIM);

    // 2) routing
    router_kernel<<<T_TOK / 8, blk256, 0, stream>>>(x, rw, rb, top_e, top_w);
    count_prefix_kernel<<<1, blk256, 0, stream>>>(top_e, ex_base, ex_cnt, cursor);
    scatter_kernel<<<(2 * T_TOK + 255) / 256, blk256, 0, stream>>>(top_e, ex_base, cursor, slot_tok, tok_slot);

    // 3) shared expert: h = gelu(x@sw1+b1) ; out = h@sw2+b2  (out written, not accumulated)
    gemm_f16_kernel<true,  true,  false, false><<<dim3(H_HID / BN, T_TOK / BM, 1), blk256, 0, stream>>>(
        xb, sw1t, sb1, nullptr, hsh, T_TOK, H_HID, D_DIM, nullptr, nullptr, nullptr);
    gemm_f16_kernel<false, false, false, false><<<dim3(D_DIM / BN, T_TOK / BM, 1), blk256, 0, stream>>>(
        hsh, sw2t, sb2, out, nullptr, T_TOK, D_DIM, H_HID, nullptr, nullptr, nullptr);

    // 4) routed experts (grouped GEMM over compact per-expert slot ranges)
    gemm_f16_kernel<true,  true,  true,  true ><<<dim3(H_HID / BN, T_TOK / BM, N_EXP), blk256, 0, stream>>>(
        xb, ew1t, eb1, nullptr, hex, 0, H_HID, D_DIM, ex_base, ex_cnt, slot_tok);
    gemm_f16_kernel<false, false, true,  false><<<dim3(D_DIM / BN, T_TOK / BM, N_EXP), blk256, 0, stream>>>(
        hex, ew2t, eb2, eout, nullptr, 0, D_DIM, H_HID, ex_base, ex_cnt, nullptr);

    // 5) combine
    combine_kernel<<<(T_TOK * D_DIM + 255) / 256, blk256, 0, stream>>>(out, eout, top_w, tok_slot);
}
